// Self_Attention_Memory_Module_75170517615300
// MI455X (gfx1250) — matrix operand order repaired and checked by reading, compiled, not run on hardware
//
#include <hip/hip_runtime.h>

// ---------------------------------------------------------------------------
// CDNA5 (gfx1250) wave32 WMMA implementation of the self-attention memory
// module.  All matrix math runs on v_wmma_f32_16x16x32_f16.
// Round 2: WMMA-ready packed weights (1 v16h load per B fragment),
//          16x64 conv tiles (4 WMMAs per A fragment), transposed V so
//          attention V fragments are vector loads.
// ---------------------------------------------------------------------------

typedef _Float16 v16h __attribute__((ext_vector_type(16)));
typedef _Float16 v8h  __attribute__((ext_vector_type(8)));
typedef float    v8f  __attribute__((ext_vector_type(8)));

union VA16 { v16h v; v8h h2[2]; _Float16 e[16]; };

static __device__ __forceinline__ v8f wmma_f16(v16h a, v16h b, v8f c) {
  // D = A(16x32,f16) * B(32x16,f16) + C(16x16,f32)
  return __builtin_amdgcn_wmma_f32_16x16x32_f16(false, a, false, b,
                                                (short)0, c, false, false);
}

#define BB 8
#define HH 48
#define WW 48
#define DD 64
#define NN (HH * WW)   /* 2304 */
#define MM (BB * NN)   /* 18432 pixel rows */

// ---------------------------------------------------------------------------
// f32 -> f16 convert with strided/offset destination.
// ---------------------------------------------------------------------------
__global__ void cvt_f32_f16(const float* __restrict__ src,
                            _Float16* __restrict__ dst,
                            int rows, int cols, int dstStride, int dstOff) {
  long long idx = (long long)blockIdx.x * blockDim.x + threadIdx.x;
  long long tot = (long long)rows * cols;
  if (idx >= tot) return;
  int r = (int)(idx / cols);
  int c = (int)(idx - (long long)r * cols);
  dst[(long long)r * dstStride + dstOff + c] = (_Float16)src[idx];
}

// ---------------------------------------------------------------------------
// Repack HWIO f32 weights into WMMA B-fragment order (f16):
//   fragment (t, kb, ct) = 512 halves at [lane*16 + h], where
//   ci = kb*32 + 16*(lane>>4) + h,  co = ct*16 + (lane&15).
// One v16h load per lane then reads a whole B fragment.
// ---------------------------------------------------------------------------
__global__ void repack_w(const float* __restrict__ src,
                         _Float16* __restrict__ dst, int Cin, int Cout) {
  int nkb = Cin >> 5, nct = Cout >> 4;
  long long tot = (long long)9 * nkb * nct * 512;
  long long idx = (long long)blockIdx.x * blockDim.x + threadIdx.x;
  if (idx >= tot) return;
  int h    = (int)(idx & 15);
  int lane = (int)((idx >> 4) & 31);
  long long fr = idx >> 9;
  int ct = (int)(fr % nct);
  long long tmp = fr / nct;
  int kb = (int)(tmp % nkb);
  int t  = (int)(tmp / nkb);
  int ci = kb * 32 + ((lane >> 4) * 16) + h;
  int co = ct * 16 + (lane & 15);
  dst[idx] = (_Float16)src[((long long)t * Cin + ci) * Cout + co];
}

// ---------------------------------------------------------------------------
// Implicit-GEMM 3x3 'SAME' conv, NHWC.  One wave -> 16 pixels x 64 out-chans
// (4 co-blocks, acc[4]); A fragment amortized over 4 WMMAs.
// Weights come pre-packed in B-fragment order.
// o16T=1 stores the f16 result transposed as [(b*Cout+co)*NN + n] (for V).
// ---------------------------------------------------------------------------
__global__ __launch_bounds__(32) void conv3_wmma(
    const _Float16* __restrict__ in, int Cin,
    const _Float16* __restrict__ wp, const float* __restrict__ bias, int Cout,
    _Float16* __restrict__ out16, int o16stride, int o16off, int o16T,
    float* __restrict__ out32) {
  const int tile = blockIdx.x;           // MM/16 tiles, each inside one row
  const int cb   = blockIdx.y;           // 64-out-channel block
  const int lane = threadIdx.x;
  const int lm   = lane & 15;
  const int hf   = lane >> 4;
  const int m0   = tile * 16;
  const int b    = m0 / NN;
  const int rem  = m0 - b * NN;
  const int y    = rem / WW;
  const int x0   = rem - y * WW;
  const int nkb  = Cin >> 5;
  const int nct  = Cout >> 4;
  const int ct0  = cb * 4;

  v8f acc[4];
#pragma unroll
  for (int t = 0; t < 4; ++t) { v8f z = {}; acc[t] = z; }

  for (int dy = 0; dy < 3; ++dy) {
    int yy = y + dy - 1;
    if (yy < 0 || yy >= HH) continue;    // wave-uniform skip
    for (int dx = 0; dx < 3; ++dx) {
      int xx = x0 + lm + dx - 1;         // per-lane shifted pixel
      bool vld = (xx >= 0 && xx < WW);
      int xs = vld ? xx : 0;
      const _Float16* inrow =
          in + ((long long)(b * NN + yy * WW + xs)) * Cin;
      int tap = dy * 3 + dx;
      for (int kb = 0; kb < nkb; ++kb) {
        // ---- A fragment (K = h + 16*hf) ----
        VA16 a;
        int ciBase = kb * 32 + (hf ? 16 : 0);
        if (vld) {
          a.h2[0] = *(const v8h*)(inrow + ciBase);
          a.h2[1] = *(const v8h*)(inrow + ciBase + 8);
        } else {
#pragma unroll
          for (int i = 0; i < 16; ++i) a.e[i] = (_Float16)0.f;
        }
        // ---- 4 pre-packed B fragments, 1 v16h load each ----
        const _Float16* wpf =
            wp + (((long long)(tap * nkb + kb) * nct + ct0) << 9) + lane * 16;
#pragma unroll
        for (int t = 0; t < 4; ++t) {
          v16h bf = *(const v16h*)(wpf + ((long long)t << 9));
          acc[t] = wmma_f16(a.v, bf, acc[t]);
        }
      }
    }
  }

#pragma unroll
  for (int t = 0; t < 4; ++t) {
    int co = ct0 * 16 + t * 16 + lm;
    float bvv = bias[co];
#pragma unroll
    for (int vv = 0; vv < 8; ++vv) {
      float val = acc[t][vv] + bvv;
      int row = vv + hf * 8;             // C layout: M = v + 8*half
      long long mrow = m0 + row;
      if (out32) out32[mrow * Cout + co] = val;
      if (out16) {
        if (o16T)
          out16[((long long)(b * Cout + co)) * NN + rem + row] = (_Float16)val;
        else
          out16[mrow * o16stride + o16off + co] = (_Float16)val;
      }
    }
  }
}

// ---------------------------------------------------------------------------
// Flash attention: softmax(Q K^T) V, no scale.  One wave per (batch, 16-row
// query tile), streaming 32-key chunks.  V supplied transposed [(b*D+d)*N+n].
// Output written f16 with col offset into the [z_h | z_m] concat buffer.
// ---------------------------------------------------------------------------
__global__ __launch_bounds__(32) void flash_attn_wmma(
    const _Float16* __restrict__ q, const _Float16* __restrict__ k,
    const _Float16* __restrict__ vT, _Float16* __restrict__ out,
    int ostride, int ooff) {
  __shared__ __align__(32) _Float16 sP[16 * 32];  // 16x32 P chunk, A layout

  const int b    = blockIdx.y;
  const int q0   = blockIdx.x * 16;
  const int lane = threadIdx.x;
  const int lm   = lane & 15;
  const int hf   = lane >> 4;
  const long long base = (long long)b * NN;

  // Q A-fragments for K-dim 0..31 and 32..63
  const _Float16* qrow = q + (base + q0 + lm) * DD;
  VA16 aq0, aq1;
  aq0.h2[0] = *(const v8h*)(qrow + (hf ? 16 : 0));
  aq0.h2[1] = *(const v8h*)(qrow + (hf ? 16 : 0) + 8);
  aq1.h2[0] = *(const v8h*)(qrow + 32 + (hf ? 16 : 0));
  aq1.h2[1] = *(const v8h*)(qrow + 32 + (hf ? 16 : 0) + 8);

  float rm[8], rl[8];
  v8f acc[4];
#pragma unroll
  for (int i = 0; i < 8; ++i) { rm[i] = -3.0e38f; rl[i] = 0.f; }
#pragma unroll
  for (int t = 0; t < 4; ++t) { v8f z = {}; acc[t] = z; }

  for (int j0 = 0; j0 < NN; j0 += 32) {
    // ---- K^T B-fragments: lane: j = lm, d = 16*hf + h (contiguous) ----
    const _Float16* k0p = k + (base + j0 + lm) * DD + hf * 16;
    const _Float16* k1p = k0p + 16 * DD;
    v16h bk00 = *(const v16h*)(k0p);        // d 0..31  (pairs aq0)
    v16h bk01 = *(const v16h*)(k0p + 32);   // d 32..63 (pairs aq1)
    v16h bk10 = *(const v16h*)(k1p);
    v16h bk11 = *(const v16h*)(k1p + 32);

    v8f s0 = {};
    s0 = wmma_f16(aq0.v, bk00, s0);
    s0 = wmma_f16(aq1.v, bk01, s0);
    v8f s1 = {};
    s1 = wmma_f16(aq0.v, bk10, s1);
    s1 = wmma_f16(aq1.v, bk11, s1);

    // ---- online softmax over 32 keys ----
    float sc[8], p0[8], p1[8];
#pragma unroll
    for (int vv = 0; vv < 8; ++vv) {
      float t = fmaxf(s0[vv], s1[vv]);
      t = fmaxf(t, __shfl_xor(t, 1, 16));
      t = fmaxf(t, __shfl_xor(t, 2, 16));
      t = fmaxf(t, __shfl_xor(t, 4, 16));
      t = fmaxf(t, __shfl_xor(t, 8, 16));
      float nm = fmaxf(rm[vv], t);
      sc[vv] = __expf(rm[vv] - nm);
      p0[vv] = __expf(s0[vv] - nm);
      p1[vv] = __expf(s1[vv] - nm);
      float rs = p0[vv] + p1[vv];
      rs += __shfl_xor(rs, 1, 16);
      rs += __shfl_xor(rs, 2, 16);
      rs += __shfl_xor(rs, 4, 16);
      rs += __shfl_xor(rs, 8, 16);
      rl[vv] = rl[vv] * sc[vv] + rs;
      rm[vv] = nm;
    }

    // ---- C-layout P -> A-layout f16 via LDS ----
    __syncthreads();
#pragma unroll
    for (int vv = 0; vv < 8; ++vv) {
      int row = vv + hf * 8;
      sP[row * 32 + lm]      = (_Float16)p0[vv];
      sP[row * 32 + 16 + lm] = (_Float16)p1[vv];
    }
    __syncthreads();
    VA16 aP;
    aP.h2[0] = *(const v8h*)(&sP[lm * 32 + (hf ? 16 : 0)]);
    aP.h2[1] = *(const v8h*)(&sP[lm * 32 + 8 + (hf ? 16 : 0)]);

    // ---- P(16x32) * V(32x64): V^T rows give contiguous key index ----
#pragma unroll
    for (int t = 0; t < 4; ++t) {
      int co = t * 16 + lm;
      v16h bv =
          *(const v16h*)(vT + ((long long)(b * DD + co)) * NN + j0 + hf * 16);
      v8f c;
#pragma unroll
      for (int vv = 0; vv < 8; ++vv) c[vv] = acc[t][vv] * sc[vv];
      acc[t] = wmma_f16(aP.v, bv, c);
    }
  }

  // epilogue: divide by row sum, write f16 into concat buffer
#pragma unroll
  for (int t = 0; t < 4; ++t) {
#pragma unroll
    for (int vv = 0; vv < 8; ++vv) {
      long long row = base + q0 + vv + hf * 8;
      out[row * ostride + ooff + t * 16 + lm] =
          (_Float16)(acc[t][vv] / rl[vv]);
    }
  }
}

// ---------------------------------------------------------------------------
// Gating epilogue: combined[.,192] -> (new_h, new_m) f32, both to d_out.
// ---------------------------------------------------------------------------
__global__ void gate_kernel(const float* __restrict__ combined,
                            const float* __restrict__ m_in,
                            float* __restrict__ out, int n) {
  int idx = blockIdx.x * blockDim.x + threadIdx.x;
  if (idx >= n) return;
  int p = idx / DD;
  int c = idx - p * DD;
  const float* row = combined + (long long)p * (3 * DD);
  float mo = row[c];
  float mg = row[DD + c];
  float mi = row[2 * DD + c];
  float mi_s = 1.f / (1.f + __expf(-mi));
  float nm = (1.f - mi_s) * m_in[idx] + mi_s * tanhf(mg);
  float nh = (1.f / (1.f + __expf(-mo))) * nm;
  out[idx] = nh;                 // new_h
  out[(long long)n + idx] = nm;  // new_m
}

// ---------------------------------------------------------------------------
// Host launcher
// ---------------------------------------------------------------------------
extern "C" void kernel_launch(void* const* d_in, const int* in_sizes, int n_in,
                              void* d_out, int out_size, void* d_ws,
                              size_t ws_size, hipStream_t stream) {
  (void)in_sizes; (void)n_in; (void)out_size; (void)ws_size;
  const float* h   = (const float*)d_in[0];
  const float* m   = (const float*)d_in[1];
  const float* Wq  = (const float*)d_in[2];
  const float* bq  = (const float*)d_in[3];
  const float* Wk  = (const float*)d_in[4];
  const float* bk  = (const float*)d_in[5];
  const float* Wk2 = (const float*)d_in[6];
  const float* bk2 = (const float*)d_in[7];
  const float* Wv  = (const float*)d_in[8];
  const float* bv  = (const float*)d_in[9];
  const float* Wv2 = (const float*)d_in[10];
  const float* bv2 = (const float*)d_in[11];
  const float* Wz  = (const float*)d_in[12];
  const float* bz  = (const float*)d_in[13];
  const float* Wm  = (const float*)d_in[14];
  const float* bm  = (const float*)d_in[15];
  float* out = (float*)d_out;

  // ---- workspace carve-up (256B aligned) ----
  char* ws = (char*)d_ws;
  size_t off = 0;
  auto alloc = [&](size_t bytes) {
    char* p = ws + off;
    off += (bytes + 255) & ~(size_t)255;
    return p;
  };
  const size_t actH = (size_t)MM * DD;                 // elements
  _Float16* hf16  = (_Float16*)alloc(actH * 2);
  _Float16* mf16  = (_Float16*)alloc(actH * 2);
  _Float16* wq16  = (_Float16*)alloc((size_t)9 * DD * DD * 2);
  _Float16* wk16  = (_Float16*)alloc((size_t)9 * DD * DD * 2);
  _Float16* wk216 = (_Float16*)alloc((size_t)9 * DD * DD * 2);
  _Float16* wv16  = (_Float16*)alloc((size_t)9 * DD * DD * 2);
  _Float16* wv216 = (_Float16*)alloc((size_t)9 * DD * DD * 2);
  _Float16* wz16  = (_Float16*)alloc((size_t)9 * 2 * DD * 2 * DD * 2);
  _Float16* wm16  = (_Float16*)alloc((size_t)9 * 3 * DD * 3 * DD * 2);
  _Float16* qf    = (_Float16*)alloc(actH * 2);
  _Float16* kf    = (_Float16*)alloc(actH * 2);
  _Float16* vT    = (_Float16*)alloc(actH * 2);        // transposed V
  _Float16* k2f   = (_Float16*)alloc(actH * 2);
  _Float16* v2T   = (_Float16*)alloc(actH * 2);        // transposed V2
  _Float16* zcat  = (_Float16*)alloc((size_t)MM * 2 * DD * 2);  // [z_h|z_m]
  _Float16* cat2  = (_Float16*)alloc((size_t)MM * 3 * DD * 2);  // [Z|h]
  float*    comb  = (float*)alloc((size_t)MM * 3 * DD * 4);

  auto cvt = [&](const float* src, _Float16* dst, int rows, int cols,
                 int stride, int o) {
    long long tot = (long long)rows * cols;
    int blocks = (int)((tot + 255) / 256);
    cvt_f32_f16<<<blocks, 256, 0, stream>>>(src, dst, rows, cols, stride, o);
  };
  auto repack = [&](const float* src, _Float16* dst, int Cin, int Cout) {
    long long tot = (long long)9 * Cin * Cout;
    int blocks = (int)((tot + 255) / 256);
    repack_w<<<blocks, 256, 0, stream>>>(src, dst, Cin, Cout);
  };

  // ---- weight repack (fused f32->f16 + B-fragment layout) ----
  repack(Wq, wq16, DD, DD);
  repack(Wk, wk16, DD, DD);
  repack(Wk2, wk216, DD, DD);
  repack(Wv, wv16, DD, DD);
  repack(Wv2, wv216, DD, DD);
  repack(Wz, wz16, 2 * DD, 2 * DD);
  repack(Wm, wm16, 3 * DD, 3 * DD);

  // ---- activation converts + h into [Z|h] concat ----
  cvt(h, hf16, MM, DD, DD, 0);
  cvt(m, mf16, MM, DD, DD, 0);
  cvt(h, cat2, MM, DD, 3 * DD, 2 * DD);  // h -> cols 128..191 of [Z|h]

  // ---- conv projections (implicit GEMM, WMMA) ----
  dim3 blk(32);
  dim3 g64(MM / 16, DD / 64);
  conv3_wmma<<<g64, blk, 0, stream>>>(hf16, DD, wq16, bq, DD,
                                      qf, DD, 0, 0, nullptr);
  conv3_wmma<<<g64, blk, 0, stream>>>(hf16, DD, wk16, bk, DD,
                                      kf, DD, 0, 0, nullptr);
  conv3_wmma<<<g64, blk, 0, stream>>>(hf16, DD, wv16, bv, DD,
                                      vT, 0, 0, 1, nullptr);   // transposed
  conv3_wmma<<<g64, blk, 0, stream>>>(mf16, DD, wk216, bk2, DD,
                                      k2f, DD, 0, 0, nullptr);
  conv3_wmma<<<g64, blk, 0, stream>>>(mf16, DD, wv216, bv2, DD,
                                      v2T, 0, 0, 1, nullptr);  // transposed
  // ---- two attentions (shared queries), outputs into [z_h|z_m] ----
  dim3 ga(NN / 16, BB);
  flash_attn_wmma<<<ga, blk, 0, stream>>>(qf, kf, vT, zcat, 2 * DD, 0);
  flash_attn_wmma<<<ga, blk, 0, stream>>>(qf, k2f, v2T, zcat, 2 * DD, DD);

  // ---- Z = conv(zcat; Wz)  -> cols 0..127 of [Z|h] ----
  dim3 g128(MM / 16, (2 * DD) / 64);
  conv3_wmma<<<g128, blk, 0, stream>>>(zcat, 2 * DD, wz16, bz, 2 * DD,
                                       cat2, 3 * DD, 0, 0, nullptr);

  // ---- combined = conv([Z|h]; Wm) -> f32 ----
  dim3 g192(MM / 16, (3 * DD) / 64);
  conv3_wmma<<<g192, blk, 0, stream>>>(cat2, 3 * DD, wm16, bm, 3 * DD,
                                       nullptr, 0, 0, 0, comb);

  // ---- gating epilogue ----
  int n = MM * DD;
  gate_kernel<<<(n + 255) / 256, 256, 0, stream>>>(comb, m, out, n);
}
